// plga_layer_81123342287393
// MI455X (gfx1250) — matrix operand order repaired and checked by reading, not compiled, not run on hardware
//
#include <hip/hip_runtime.h>

#define B_ 2
#define H_ 16
#define S_ 2048
#define D_ 64
#define QT 128
#define KTILE 64
#define KPAD 72   /* f16 row stride for K/V tiles: 144B, 16B-aligned, bank-friendly */

typedef __attribute__((ext_vector_type(16))) _Float16 v16h;
typedef __attribute__((ext_vector_type(8)))  _Float16 v8h;
typedef __attribute__((ext_vector_type(8)))  float    v8f;
typedef __attribute__((ext_vector_type(4)))  unsigned u32x4;
typedef __attribute__((ext_vector_type(8)))  unsigned u32x8;

__device__ __forceinline__ v8f wmma_f16(v16h a, v16h b, v8f c) {
  // D = A(16x32 f16) * B(32x16 f16) + C(16x16 f32)
  return __builtin_amdgcn_wmma_f32_16x16x32_f16(false, a, false, b, (short)0, c,
                                                false, false);
}

__device__ __forceinline__ v16h cat16(v8h a, v8h b) {
  return __builtin_shufflevector(a, b, 0,1,2,3,4,5,6,7,8,9,10,11,12,13,14,15);
}
// 16 contiguous f16 (32B, 16B-aligned) -> v16h via two b128 LDS/global reads
__device__ __forceinline__ v16h load_frag16(const _Float16* p) {
  return cat16(*(const v8h*)p, *(const v8h*)(p + 8));
}
// 8 contiguous f32 -> 8 f16 (optionally pre-scaled)
__device__ __forceinline__ v8h cvt8(const float* p) {
  v8h r;
  #pragma unroll
  for (int t = 0; t < 8; ++t) r[t] = (_Float16)p[t];
  return r;
}
__device__ __forceinline__ v8h cvt8s(const float* p, float s) {
  v8h r;
  #pragma unroll
  for (int t = 0; t < 8; ++t) r[t] = (_Float16)(p[t] * s);
  return r;
}

// ---------------------------------------------------------------------------
// Tensor Data Mover: 2D tile (64 x 64 x 2B) global -> LDS, with LDS padding
// of +4 DWORDs every 32 DWORDs (row 128B -> 144B = KPAD*2).
// D# per CDNA5 ISA.  Issued by one wave; tracked
// by TENSORcnt.
// ---------------------------------------------------------------------------
__device__ __forceinline__ void tdm_load_tile(unsigned lds_addr, const void* gbase,
                                              unsigned tensor_d0, unsigned tensor_d1,
                                              unsigned stride0)
{
  const unsigned long long ga = (unsigned long long)(size_t)gbase;
  u32x4 g0;
  g0[0] = 1u;                                              // count=1 (valid), user mode
  g0[1] = lds_addr;                                        // lds_addr[31:0]
  g0[2] = (unsigned)(ga & 0xFFFFFFFFu);                    // global_addr[31:0]
  g0[3] = (unsigned)((ga >> 32) & 0x1FFFFFFu) | (2u << 30); // addr[56:32] | type=2
  u32x8 g1;
  g1[0] = (1u << 16)        // data_size = 1 -> 2 bytes
        | (1u << 20)        // pad_enable
        | (4u << 22)        // pad_interval: 32 DWORDs
        | (3u << 25);       // pad_amount: 4 DWORDs
  g1[1] = (tensor_d0 & 0xFFFFu) << 16;                     // tensor_dim0[15:0] @63:48
  g1[2] = (tensor_d0 >> 16) | ((tensor_d1 & 0xFFFFu) << 16); // dim0 hi | dim1 lo
  g1[3] = (tensor_d1 >> 16) | (64u << 16);                 // dim1 hi | tile_dim0=64
  g1[4] = 64u;                                             // tile_dim1=64 | tile_dim2=0
  g1[5] = stride0;                                         // tensor_dim0_stride[31:0]
  g1[6] = 0u;                                              // stride0 hi | stride1 lo
  g1[7] = 0u;
  asm volatile("tensor_load_to_lds %0, %1" :: "s"(g0), "s"(g1) : "memory");
}

// ---------------------------------------------------------------------------
// Kernel 1: per (b,h):
//   avAp = a @ pow(iswiglu(W@A+b)+eps, pw) + ba          (fp32, LDS)
//   Kt   = Hk @ avAp^T   (f16 -> ws)  via WMMA
//   Hvt  = Hv^T          (f16 -> ws)  LDS-tiled transpose
// ---------------------------------------------------------------------------
__global__ __launch_bounds__(256, 1)
void prep_kt_kernel(const float* __restrict__ Hk, const float* __restrict__ Hv,
                    const float* __restrict__ A,  const float* __restrict__ W,
                    const float* __restrict__ bias, const float* __restrict__ pw,
                    const float* __restrict__ aM, const float* __restrict__ ba,
                    _Float16* __restrict__ Kt, _Float16* __restrict__ Hvt)
{
  __shared__ __align__(16) float    s_w[D_*D_];
  __shared__ __align__(16) float    s_a[D_*D_];
  __shared__ __align__(16) float    s_x[D_*D_];
  __shared__ __align__(16) _Float16 s_avh[D_*D_];
  __shared__ __align__(16) _Float16 s_t[D_][65];

  const int bh  = blockIdx.x;
  const int h   = bh % H_;
  const int tid = threadIdx.x;
  const int lane = tid & 31, wave = tid >> 5;
  const int ln = lane & 15, lh = lane >> 4;

  const float* Wp = W + (size_t)h*D_*D_;
  const float* Ap = A + (size_t)bh*D_*D_;
  for (int i = tid; i < D_*D_; i += 256) { s_w[i] = Wp[i]; s_a[i] = Ap[i]; }
  __syncthreads();

  // x = W@A + b ; Ap = pow(iswiglu(x)+eps, pw)
  for (int idx = tid; idx < D_*D_; idx += 256) {
    const int i = idx >> 6, j = idx & 63;
    float acc = bias[(size_t)h*D_*D_ + idx];
    for (int k = 0; k < D_; ++k) acc = fmaf(s_w[i*D_+k], s_a[k*D_+j], acc);
    const float sig = 1.0f / (1.0f + __expf(-acc));
    const float AW  = acc*acc*sig + 1e-9f;                 // x*silu(x) + eps > 0
    s_x[idx] = __expf(pw[(size_t)h*D_*D_ + idx] * __logf(AW));
  }
  __syncthreads();

  // avAp = a @ Ap + ba  ->  f16 LDS
  for (int idx = tid; idx < D_*D_; idx += 256) {
    const int i = idx >> 6, j = idx & 63;
    float acc = ba[(size_t)h*D_*D_ + idx];
    for (int k = 0; k < D_; ++k) acc = fmaf(aM[(size_t)h*D_*D_ + i*D_+k], s_x[k*D_+j], acc);
    s_avh[idx] = (_Float16)acc;
  }
  __syncthreads();

  // Hoisted B fragments of avAp^T: B[k][n] = avAp[n][k]
  v16h Bf[4][2];
  #pragma unroll
  for (int nb = 0; nb < 4; ++nb)
    #pragma unroll
    for (int kb = 0; kb < 2; ++kb)
      Bf[nb][kb] = load_frag16(&s_avh[(nb*16 + ln)*D_ + kb*32 + lh*16]);

  const float* HkB = Hk + (size_t)bh*S_*D_;
  _Float16*    KtB = Kt + (size_t)bh*S_*D_;

  for (int st = 0; st < S_/128; ++st) {
    const int sBase = st*128 + wave*16;
    const float* rowp = HkB + (size_t)(sBase + ln)*D_;
    // A layout: lane<16 row=ln, K={0..15}+32f ; lane>=16 K={16..31}+32f
    v16h Af[2];
    #pragma unroll
    for (int f = 0; f < 2; ++f) {
      const int c0 = f*32 + lh*16;
      Af[f] = cat16(cvt8(rowp + c0), cvt8(rowp + c0 + 8));
    }
    #pragma unroll
    for (int nb = 0; nb < 4; ++nb) {
      v8f c = {};
      c = wmma_f16(Af[0], Bf[nb][0], c);
      c = wmma_f16(Af[1], Bf[nb][1], c);
      #pragma unroll
      for (int r = 0; r < 8; ++r) {
        const int row = sBase + (lh ? 8 + r : r);
        KtB[(size_t)row*D_ + nb*16 + ln] = (_Float16)c[r];
      }
    }
  }

  // ---- Hv^T (f16) via LDS-tiled transpose ----
  const float* HvB  = Hv  + (size_t)bh*S_*D_;
  _Float16*    HvtB = Hvt + (size_t)bh*S_*D_;
  for (int st2 = 0; st2 < S_/64; ++st2) {
    const int sBase = st2*64;
    for (int idx = tid; idx < 64*64; idx += 256) {
      const int s = idx >> 6, d = idx & 63;
      s_t[d][s] = (_Float16)HvB[(size_t)(sBase + s)*D_ + d];
    }
    __syncthreads();
    for (int idx = tid; idx < 64*64; idx += 256) {
      const int d = idx >> 6, s = idx & 63;
      HvtB[(size_t)d*S_ + sBase + s] = s_t[d][s];
    }
    __syncthreads();
  }
}

// ---------------------------------------------------------------------------
// Kernel 2: causal flash attention, double-buffered TDM K/V staging.
// Q = Hin (f16*scale on load), K = Kt (ws f16), V^T = Hvt (ws f16).
// Block = (b,h, 128-query tile); wave = 16 query rows.
// ---------------------------------------------------------------------------
__global__ __launch_bounds__(256, 1)
void attn_kernel(const float* __restrict__ Hin, const _Float16* __restrict__ Kt,
                 const _Float16* __restrict__ Hvt, float* __restrict__ Hout)
{
  __shared__ __align__(16) _Float16 s_k [2][KTILE][KPAD];  // keys x d
  __shared__ __align__(16) _Float16 s_vt[2][KTILE][KPAD];  // d x keys
  __shared__ __align__(16) _Float16 s_p [8][16][KPAD];     // per-wave P scratch

  const int nQT = S_/QT;
  const int bh  = blockIdx.x / nQT;
  const int qt  = blockIdx.x % nQT;
  const int tid = threadIdx.x, lane = tid & 31, wave = tid >> 5;
  const int ln = lane & 15, lh = lane >> 4;

  const float*    HinB = Hin + (size_t)bh*S_*D_;
  const _Float16* KtB  = Kt  + (size_t)bh*S_*D_;
  const _Float16* HvtB = Hvt + (size_t)bh*S_*D_;
  float*          HoB  = Hout + (size_t)bh*S_*D_;

  const int qBase = qt*QT + wave*16;

  // Q A-fragments, pre-scaled by 1/sqrt(D)
  v16h Qf[2];
  {
    const float* rowp = HinB + (size_t)(qBase + ln)*D_;
    #pragma unroll
    for (int f = 0; f < 2; ++f) {
      const int c0 = f*32 + lh*16;
      Qf[f] = cat16(cvt8s(rowp + c0, 0.125f), cvt8s(rowp + c0 + 8, 0.125f));
    }
  }

  float m_run[8], l_run[8];
  v8f   Oacc[4];
  #pragma unroll
  for (int r = 0; r < 8; ++r) { m_run[r] = -1e30f; l_run[r] = 0.0f; }
  #pragma unroll
  for (int nb = 0; nb < 4; ++nb) { v8f z = {}; Oacc[nb] = z; }

  const int kBlocks = (qt + 1) * (QT/KTILE);   // causal key-tile bound

  // ---- TDM staging: wave 0 issues one descriptor per tile ----
  auto stage = [&](int kStart, int buf) {
    tdm_load_tile((unsigned)(size_t)&s_k[buf][0][0],
                  KtB + (size_t)kStart*D_, /*d0*/D_, /*d1*/S_, /*stride*/D_);
    tdm_load_tile((unsigned)(size_t)&s_vt[buf][0][0],
                  HvtB + kStart, /*d0*/S_, /*d1*/D_, /*stride*/S_);
  };

  if (wave == 0) stage(0, 0);
  for (int kb = 0; kb < kBlocks; ++kb) {
    const int kStart = kb*KTILE;
    const int cur = kb & 1;
    if (wave == 0) {
      if (kb + 1 < kBlocks) {
        stage(kStart + KTILE, cur ^ 1);
        __builtin_amdgcn_s_wait_tensorcnt(2);   // kb's 2 TDM ops complete (in-order)
      } else {
        __builtin_amdgcn_s_wait_tensorcnt(0);
      }
    }
    __syncthreads();

    if (kStart <= qBase + 15) {                  // wave has unmasked keys here
      // ---- scores: S = (Q/sqrtD) * Kt^T   (4 tiles of 16 keys) ----
      v8f Sc[4];
      #pragma unroll
      for (int nb = 0; nb < 4; ++nb) {
        v8f c = {};
        #pragma unroll
        for (int f = 0; f < 2; ++f) {
          // B[k][n] = Kt[key n][k]: contiguous row read
          v16h v = load_frag16(&s_k[cur][nb*16 + ln][f*32 + lh*16]);
          c = wmma_f16(Qf[f], v, c);
        }
        Sc[nb] = c;
      }
      // ---- causal mask (select only; scale folded into Q) ----
      #pragma unroll
      for (int nb = 0; nb < 4; ++nb) {
        const int key = kStart + nb*16 + ln;
        #pragma unroll
        for (int r = 0; r < 8; ++r) {
          const int qrow = qBase + (lh ? 8 + r : r);
          Sc[nb][r] = (key <= qrow) ? Sc[nb][r] : -1e30f;
        }
      }
      // ---- online softmax (row reductions within 16-lane half-groups) ----
      #pragma unroll
      for (int r = 0; r < 8; ++r) {
        float mx = fmaxf(fmaxf(Sc[0][r], Sc[1][r]), fmaxf(Sc[2][r], Sc[3][r]));
        mx = fmaxf(mx, __shfl_xor(mx, 1, 32));
        mx = fmaxf(mx, __shfl_xor(mx, 2, 32));
        mx = fmaxf(mx, __shfl_xor(mx, 4, 32));
        mx = fmaxf(mx, __shfl_xor(mx, 8, 32));
        const float mnew  = fmaxf(m_run[r], mx);
        const float alpha = __expf(m_run[r] - mnew);
        m_run[r] = mnew;

        const int prow = (lh ? 8 + r : r);
        float psum = 0.0f;
        #pragma unroll
        for (int nb = 0; nb < 4; ++nb) {
          const float p = __expf(Sc[nb][r] - mnew);
          psum += p;
          s_p[wave][prow][nb*16 + ln] = (_Float16)p;
        }
        psum += __shfl_xor(psum, 1, 32);
        psum += __shfl_xor(psum, 2, 32);
        psum += __shfl_xor(psum, 4, 32);
        psum += __shfl_xor(psum, 8, 32);
        l_run[r] = l_run[r]*alpha + psum;
        #pragma unroll
        for (int nb = 0; nb < 4; ++nb) Oacc[nb][r] *= alpha;
      }
      // cross-lane ds-store -> ds-load hazard inside the wave
      asm volatile("s_wait_dscnt 0" ::: "memory");

      // ---- O += P * V   (K = 64 keys in two K=32 chunks) ----
      #pragma unroll
      for (int kc = 0; kc < 2; ++kc) {
        const _Float16* prow = &s_p[wave][ln][0];
        const int c0 = kc*32 + lh*16;
        v16h Pf = cat16(*(const v8h*)(prow + c0), *(const v8h*)(prow + c0 + 8));
        #pragma unroll
        for (int nb = 0; nb < 4; ++nb) {
          // B[k][n] = V[kStart+kc*32+k][n] = Hvt row read: contiguous
          v16h v = load_frag16(&s_vt[cur][nb*16 + ln][kc*32 + lh*16]);
          Oacc[nb] = wmma_f16(Pf, v, Oacc[nb]);
        }
      }
    }
    __syncthreads();
  }

  // ---- normalize + store ----
  #pragma unroll
  for (int r = 0; r < 8; ++r) {
    const float inv = 1.0f / l_run[r];
    const int row = qBase + (lh ? 8 + r : r);
    #pragma unroll
    for (int nb = 0; nb < 4; ++nb) {
      HoB[(size_t)row*D_ + nb*16 + ln] = Oacc[nb][r] * inv;
    }
  }
}

// ---------------------------------------------------------------------------
extern "C" void kernel_launch(void* const* d_in, const int* in_sizes, int n_in,
                              void* d_out, int out_size, void* d_ws, size_t ws_size,
                              hipStream_t stream)
{
  const float* Hin  = (const float*)d_in[0];
  const float* Hk   = (const float*)d_in[1];
  const float* Hv   = (const float*)d_in[2];
  const float* A    = (const float*)d_in[3];
  // d_in[4] = mask: unused, causality computed analytically
  const float* W    = (const float*)d_in[5];
  const float* bias = (const float*)d_in[6];
  const float* pw   = (const float*)d_in[7];
  const float* aM   = (const float*)d_in[8];
  const float* ba   = (const float*)d_in[9];
  float* Hout = (float*)d_out;

  const size_t elems = (size_t)B_*H_*S_*D_;
  _Float16* Kt  = (_Float16*)d_ws;
  _Float16* Hvt = Kt + elems;
  if (ws_size < 2*elems*sizeof(_Float16)) return;   // 16 MB workspace

  prep_kt_kernel<<<B_*H_, 256, 0, stream>>>(Hk, Hv, A, W, bias, pw, aM, ba, Kt, Hvt);
  attn_kernel<<<B_*H_*(S_/QT), 256, 0, stream>>>(Hin, Kt, Hvt, Hout);
}
